// QCformer_54254026883838
// MI455X (gfx1250) — matrix operand order repaired and checked by reading, not compiled, not run on hardware
//
#include <hip/hip_runtime.h>
#include <hip/hip_bf16.h>

// ---------------------------------------------------------------------------
// QCformer forward for MI455X (gfx1250, wave32, WMMA).
// All dense GEMMs: f32 -> bf16, staged into LDS in WMMA *fragment order* so
// compute does wide ds_load_b128 reads, then v_wmma_f32_16x16x32_bf16.
// Main K-loop is guard-free; K tails handled by a single masked tail chunk.
// ---------------------------------------------------------------------------

typedef __attribute__((ext_vector_type(16))) __bf16 v16bf;
typedef __attribute__((ext_vector_type(4)))  __bf16 v4bf;
typedef __attribute__((ext_vector_type(8)))  float  v8f;

#define LRELU(v) ((v) > 0.f ? (v) : 0.01f * (v))
#define EPSLN 1e-5f

__device__ __forceinline__ int imin(int a, int b) { return a < b ? a : b; }

// ---------------- generic WMMA GEMM ----------------------------------------
// C[M,N] = epi(A[M,K] @ W[K,N] + bias)
// epi: 0 = none, 1 = multiply by aux[row*ldaux+col] (gate), 2 = leaky_relu
// Block: 256 threads = 8 waves. Block tile 32(M) x 128(N). Wave tile 16x32.
//
// LDS holds tiles in WMMA fragment order:
//   Af[t][lane][e]  : A 16x32 bf16 fragment for M-tile t (t = 0,1)
//      lane 0-15  -> M = lane,    K = e       (e = 0..15)
//      lane 16-31 -> M = lane-16, K = 16 + e
//   Bf[s][lane][e]  : B 32x16 bf16 fragment for N-subtile s (s = 0..7)
//      lane 0-15  -> N = lane,    K = e       (e = 0..15)
//      lane 16-31 -> N = lane-16, K = 16 + e
__global__ __launch_bounds__(256) void wmma_gemm_kernel(
    const float* __restrict__ A, int M, int K, int lda,
    const float* __restrict__ W, int ldw,
    const float* __restrict__ bias,
    float* __restrict__ C, int ldc,
    const float* __restrict__ aux, int ldaux, int epi)
{
  __shared__ alignas(32) __bf16 Af[2][32][16];   // 2 KB
  __shared__ alignas(32) __bf16 Bf[8][32][16];   // 8 KB

  const int tid  = threadIdx.x;
  const int lane = tid & 31;
  const int wave = tid >> 5;
  const int wm   = wave & 1;       // wave row: 0..1  -> 16 rows each
  const int wn   = wave >> 1;      // wave col: 0..3  -> 32 cols each
  const int m0   = blockIdx.y * 32;
  const int n0   = blockIdx.x * 128;

  v8f acc0 = {0.f,0.f,0.f,0.f,0.f,0.f,0.f,0.f};
  v8f acc1 = {0.f,0.f,0.f,0.f,0.f,0.f,0.f,0.f};

  // --- staging thread maps (fixed per thread) ------------------------------
  // A: 64 lane-vectors (2 tiles x 32 lanes), 4 threads per vector; each
  // thread's 4 fragment elems are 4 *consecutive* K values -> one float4.
  const int avec  = tid >> 2;            // 0..63
  const int apart = tid & 3;             // elems apart*4 .. apart*4+3
  const int awm   = avec >> 5;           // M tile 0/1
  const int aln   = avec & 31;           // target lane
  const int am    = aln & 15;            // M within tile
  const int ahi8  = (aln & 16) ? 8 : 0;  // hi lanes: K offset 2*ahi8 = 16
  const int kbase = apart * 4 + 2 * ahi8;
  // B: 256 lane-vectors (8 subtiles x 32 lanes), 1 thread per vector.
  const int bcol  = tid & 127;           // N within 128
  const int bkh   = tid >> 7;            // K half: 0 -> K 0..15, 1 -> 16..31
  const int bsub  = bcol >> 4;
  const int bln   = (bcol & 15) + (bkh ? 16 : 0);

  const int arow    = m0 + awm * 16 + am;
  const int arcl    = imin(arow, M - 1);
  const float amask = (arow < M) ? 1.f : 0.f;

  const float* Ap = A + (long long)arcl * lda;                   // row base
  const float* Wp = W + (long long)(bkh * 16) * ldw + (n0 + bcol);

  const int Kfull = K & ~31;

  // one fragment-load + dual-WMMA step (tiles already staged in LDS)
  auto mma_step = [&]() {
    __syncthreads();
    v16bf a  = *(const v16bf*)&Af[wm][lane][0];          // 2x ds_load_b128
    v16bf b0 = *(const v16bf*)&Bf[wn * 2 + 0][lane][0];
    v16bf b1 = *(const v16bf*)&Bf[wn * 2 + 1][lane][0];
#if defined(__HIP_DEVICE_COMPILE__)
    acc0 = __builtin_amdgcn_wmma_f32_16x16x32_bf16(false, a, false, b0,
                                                   (short)0, acc0, false, false);
    acc1 = __builtin_amdgcn_wmma_f32_16x16x32_bf16(false, a, false, b1,
                                                   (short)0, acc1, false, false);
#endif
    __syncthreads();
  };

  int k0 = 0;
  for (; k0 < Kfull; k0 += 32) {
    // ---- stage A: one 16B load, mask-multiplied (branchless) ----
    {
      float4 a4 = *(const float4*)(Ap + k0 + kbase);
      v4bf av;
      av[0] = (__bf16)(a4.x * amask);
      av[1] = (__bf16)(a4.y * amask);
      av[2] = (__bf16)(a4.z * amask);
      av[3] = (__bf16)(a4.w * amask);
      *(v4bf*)&Af[awm][aln][apart * 4] = av;             // ds_store_b64
    }
    // ---- stage B: 16 coalesced column loads, no guards ----
    {
      v16bf bv;
      #pragma unroll
      for (int j = 0; j < 16; ++j)
        bv[j] = (__bf16)Wp[(long long)j * ldw];
      *(v16bf*)&Bf[bsub][bln][0] = bv;                   // 2x ds_store_b128
      Wp += (long long)32 * ldw;
    }
    mma_step();
  }

  // ---- K tail (only for K % 32 != 0: the input projections) ----
  if (k0 < K) {
    {
      v4bf av;
      #pragma unroll
      for (int j = 0; j < 4; ++j) {
        int kg = k0 + kbase + j;
        float mk = (kg < K) ? amask : 0.f;
        av[j] = (__bf16)(Ap[imin(kg, K - 1)] * mk);
      }
      *(v4bf*)&Af[awm][aln][apart * 4] = av;
    }
    {
      v16bf bv;
      #pragma unroll
      for (int j = 0; j < 16; ++j) {
        int kg = k0 + bkh * 16 + j;
        float mk = (kg < K) ? 1.f : 0.f;
        bv[j] = (__bf16)(W[(long long)imin(kg, K - 1) * ldw + n0 + bcol] * mk);
      }
      *(v16bf*)&Bf[bsub][bln][0] = bv;
    }
    mma_step();
  }

  // epilogue: C/D layout -> VGPR r holds M=r (lanes 0-15) / M=8+r (lanes 16-31)
  const int ncol0 = wn * 32 + (lane & 15);
  const int ncol1 = ncol0 + 16;
  const int rbase = m0 + wm * 16 + ((lane < 16) ? 0 : 8);
  #pragma unroll
  for (int r = 0; r < 8; ++r) {
    int row = rbase + r;
    if (row < M) {
      int c0 = n0 + ncol0, c1 = n0 + ncol1;
      float v0 = acc0[r] + (bias ? bias[c0] : 0.f);
      float v1 = acc1[r] + (bias ? bias[c1] : 0.f);
      if (epi == 1) {
        v0 *= aux[(long long)row * ldaux + c0];
        v1 *= aux[(long long)row * ldaux + c1];
      } else if (epi == 2) {
        v0 = LRELU(v0); v1 = LRELU(v1);
      }
      C[(long long)row * ldc + c0] = v0;
      C[(long long)row * ldc + c1] = v1;
    }
  }
}

// ---------------- elementwise / graph kernels -------------------------------

__global__ void zero_f32(float* __restrict__ p, long long n) {
  long long i = (long long)blockIdx.x * 256 + threadIdx.x;
  long long stride = (long long)gridDim.x * 256;
  for (; i < n; i += stride) p[i] = 0.f;
}

// per-edge: alpha = cat(q*k_j, q*KE)*scale; gate = sigmoid(LN(alpha,g1,b1));
// m = cat(v_j, VE).   KV = [Kv | Vv] (n x 256), KE = [Ke | Ve] (e x 256).
__global__ __launch_bounds__(256) void edge_gate_kernel(
    const float* __restrict__ KV, const float* __restrict__ KE,
    const int* __restrict__ src, const int* __restrict__ dst,
    const float* __restrict__ g1, const float* __restrict__ b1,
    float* __restrict__ GATE, float* __restrict__ MB)
{
  __shared__ float s1[256];
  __shared__ float s2[256];
  const int ed = blockIdx.x;
  const int c  = threadIdx.x;
  const long long sb = (long long)src[ed] * 256;
  const long long db = (long long)dst[ed] * 256;
  const long long eb = (long long)ed * 256;

  float q  = KV[db + (c & 127)];
  float kk = (c < 128) ? KV[sb + c] : KE[eb + (c - 128)];
  float alpha = q * kk * 0.0625f;               // 1/sqrt(2*128)

  s1[c] = alpha; s2[c] = alpha * alpha;
  __syncthreads();
  for (int o = 128; o > 0; o >>= 1) {
    if (c < o) { s1[c] += s1[c + o]; s2[c] += s2[c + o]; }
    __syncthreads();
  }
  float mu  = s1[0] * (1.f / 256.f);
  float var = s2[0] * (1.f / 256.f) - mu * mu;
  float t = (alpha - mu) * rsqrtf(var + EPSLN) * g1[c] + b1[c];
  GATE[eb + c] = 1.f / (1.f + __expf(-t));

  float mv = (c < 128) ? KV[sb + 128 + c] : KE[eb + 128 + (c - 128)];
  MB[eb + c] = mv;
}

// msg = lrelu(LN(t, g2, b2)); atomically scatter into HV[dst, hoff:hoff+128]
__global__ __launch_bounds__(128) void ln_scatter_kernel(
    const float* __restrict__ T, const float* __restrict__ g2,
    const float* __restrict__ b2, const int* __restrict__ dst,
    float* __restrict__ HV, int hoff)
{
  __shared__ float s1[128];
  __shared__ float s2[128];
  const int ed = blockIdx.x;
  const int c  = threadIdx.x;
  float t = T[(long long)ed * 128 + c];
  s1[c] = t; s2[c] = t * t;
  __syncthreads();
  for (int o = 64; o > 0; o >>= 1) {
    if (c < o) { s1[c] += s1[c + o]; s2[c] += s2[c + o]; }
    __syncthreads();
  }
  float mu  = s1[0] * (1.f / 128.f);
  float var = s2[0] * (1.f / 128.f) - mu * mu;
  float v = (t - mu) * rsqrtf(var + EPSLN) * g2[c] + b2[c];
  v = LRELU(v);
  atomicAdd(&HV[(long long)dst[ed] * 512 + hoff + c], v);
}

// per-column partial sums for BatchNorm (biased var)
__global__ __launch_bounds__(128) void bn_stats_kernel(
    const float* __restrict__ X, int n,
    float* __restrict__ S, float* __restrict__ S2)
{
  const int c  = threadIdx.x;
  const int r0 = blockIdx.x * 256;
  int r1 = r0 + 256; if (r1 > n) r1 = n;
  float s = 0.f, q = 0.f;
  for (int r = r0; r < r1; ++r) {
    float v = X[(long long)r * 128 + c];
    s += v; q += v * v;
  }
  atomicAdd(&S[c], s);
  atomicAdd(&S2[c], q);
}

// feat += lrelu(BN(out))
__global__ __launch_bounds__(256) void bn_apply_kernel(
    const float* __restrict__ O, float* __restrict__ feat, int n,
    const float* __restrict__ S, const float* __restrict__ S2,
    const float* __restrict__ g, const float* __restrict__ b)
{
  long long i = (long long)blockIdx.x * 256 + threadIdx.x;
  if (i >= (long long)n * 128) return;
  int c = (int)(i & 127);
  float inv_n = 1.f / (float)n;
  float mu  = S[c] * inv_n;
  float var = S2[c] * inv_n - mu * mu;
  float v = (O[i] - mu) * rsqrtf(var + EPSLN) * g[c] + b[c];
  feat[i] += LRELU(v);
}

__global__ __launch_bounds__(128) void pool_scatter_kernel(
    const float* __restrict__ X, const int* __restrict__ seg,
    float* __restrict__ P, float* __restrict__ CNT)
{
  const int r = blockIdx.x;
  const int c = threadIdx.x;
  const int bz = seg[r];
  atomicAdd(&P[(long long)bz * 128 + c], X[(long long)r * 128 + c]);
  if (c == 0) atomicAdd(&CNT[bz], 1.f);
}

__global__ __launch_bounds__(256) void pool_final_kernel(
    const float* __restrict__ P1, const float* __restrict__ C1,
    const float* __restrict__ P2, const float* __restrict__ C2,
    float* __restrict__ F)
{
  const int bz = blockIdx.x;
  const int c  = threadIdx.x;
  float v;
  if (c < 128) v = P1[bz * 128 + c] / fmaxf(C1[bz], 1.f);
  else         v = P2[bz * 128 + (c - 128)] / fmaxf(C2[bz], 1.f);
  F[bz * 256 + c] = v;
}

__global__ __launch_bounds__(128) void out_dot_kernel(
    const float* __restrict__ G, const float* __restrict__ W,
    const float* __restrict__ b, float* __restrict__ out)
{
  __shared__ float red[128];
  const int bz = blockIdx.x;
  const int c  = threadIdx.x;
  red[c] = G[bz * 128 + c] * W[c];
  __syncthreads();
  for (int o = 64; o > 0; o >>= 1) {
    if (c < o) red[c] += red[c + o];
    __syncthreads();
  }
  if (c == 0) out[bz] = red[0] + b[0];
}

// ---------------- host-side orchestration ----------------------------------

namespace {

struct Head {
  const float *Ke, *Kv, *Ve, *Vv, *Wm, *Wu, *b1, *b2, *bm, *bu, *g1, *g2;
};
struct Conv {
  const float *Wc, *bc, *bn_b, *bn_g;
  Head h[4];
};

inline void gemm(hipStream_t s, const float* A, int M, int K, int lda,
                 const float* W, int ldw, const float* bias,
                 float* C, int ldc, int N,
                 const float* aux = nullptr, int ldaux = 0, int epi = 0)
{
  dim3 g(N / 128, (M + 31) / 32);
  wmma_gemm_kernel<<<g, dim3(256), 0, s>>>(A, M, K, lda, W, ldw, bias,
                                           C, ldc, aux, ldaux, epi);
}

inline void zero(hipStream_t s, float* p, long long n) {
  long long blk = (n + 255) / 256;
  if (blk > 4096) blk = 4096;
  zero_f32<<<dim3((unsigned)blk), dim3(256), 0, s>>>(p, n);
}

// one QCConv: feat[n,128] updated in place; graph edges (src,dst) with edge
// features efeat[e,128].
void run_conv(hipStream_t s, float* feat, int n,
              const int* src, const int* dst,
              const float* efeat, int e, const Conv& cv,
              float* KV, float* KE, float* GATE, float* MB,
              float* HV, float* OUTB, float* S, float* S2)
{
  zero(s, HV, (long long)n * 512);
  for (int h = 0; h < 4; ++h) {
    const Head& H = cv.h[h];
    // KV = [feat@Kv | feat@Vv]  (n x 256)
    gemm(s, feat, n, 128, 128, H.Kv, 128, nullptr, KV,       256, 128);
    gemm(s, feat, n, 128, 128, H.Vv, 128, nullptr, KV + 128, 256, 128);
    // KE = [efeat@Ke | efeat@Ve]  (e x 256)
    gemm(s, efeat, e, 128, 128, H.Ke, 128, nullptr, KE,       256, 128);
    gemm(s, efeat, e, 128, 128, H.Ve, 128, nullptr, KE + 128, 256, 128);
    // gate + message inputs
    edge_gate_kernel<<<dim3(e), dim3(256), 0, s>>>(KV, KE, src, dst,
                                                   H.g1, H.b1, GATE, MB);
    // h = (m @ Wu + bu) * gate  -> reuse KE buffer (e x 256)
    gemm(s, MB, e, 256, 256, H.Wu, 256, H.bu, KE, 256, 256, GATE, 256, 1);
    // t = h @ Wm + bm  (e x 128) -> reuse MB buffer
    gemm(s, KE, e, 256, 256, H.Wm, 128, H.bm, MB, 128, 128);
    // msg = lrelu(LN(t)); segment_sum into HV[:, h*128 : h*128+128]
    ln_scatter_kernel<<<dim3(e), dim3(128), 0, s>>>(MB, H.g2, H.b2, dst,
                                                    HV, h * 128);
  }
  // out = HV @ Wc + bc   (n x 128)
  gemm(s, HV, n, 512, 512, cv.Wc, 128, cv.bc, OUTB, 128, 128);
  // BatchNorm (training mode, biased var) + lrelu + residual
  zero(s, S, 128);
  zero(s, S2, 128);
  bn_stats_kernel<<<dim3((n + 255) / 256), dim3(128), 0, s>>>(OUTB, n, S, S2);
  long long tot = (long long)n * 128;
  bn_apply_kernel<<<dim3((unsigned)((tot + 255) / 256)), dim3(256), 0, s>>>(
      OUTB, feat, n, S, S2, cv.bn_g, cv.bn_b);
}

} // namespace

extern "C" void kernel_launch(void* const* d_in, const int* in_sizes, int n_in,
                              void* d_out, int out_size, void* d_ws, size_t ws_size,
                              hipStream_t stream)
{
  (void)in_sizes; (void)n_in; (void)out_size; (void)ws_size;
  const int NN = 12000, EE = 100000, TT = 200000;

  const float* x    = (const float*)d_in[0];   // [N, 92]
  const float* edis = (const float*)d_in[1];   // [E, 376]
  const float* tdis = (const float*)d_in[2];   // [T, 516]
  const int*   eidx = (const int*)d_in[3];     // [2, E]
  const int*   tidx = (const int*)d_in[4];     // [2, T]
  const int*   xb   = (const int*)d_in[5];     // [N]
  const int*   ebat = (const int*)d_in[6];     // [E]

  // params flattened in JAX pytree order (dict keys sorted, ASCII):
  // atom_W, atom_b, edge_W, edge_b, fc2_W, fc2_b, fc_W, fc_b,
  // layer1 (2 convs x 52 leaves), layer2 (2 x 52), out_W, out_b, tri_W, tri_b
  // conv: Wc, bc, bn_b, bn_g, heads[4] x {Ke,Kv,Ve,Vv,Wm,Wu,b1,b2,bm,bu,g1,g2}
  auto pf = [&](int i) -> const float* { return (const float*)d_in[7 + i]; };
  const float* atom_W = pf(0);  const float* atom_b = pf(1);
  const float* edge_W = pf(2);  const float* edge_b = pf(3);
  const float* fc2_W  = pf(4);  const float* fc2_b  = pf(5);
  const float* fc_W   = pf(6);  const float* fc_b   = pf(7);
  const float* out_W  = pf(216); const float* out_b = pf(217);
  const float* tri_W  = pf(218); const float* tri_b = pf(219);

  auto mk_conv = [&](int base) -> Conv {
    Conv cv;
    cv.Wc = pf(base); cv.bc = pf(base + 1);
    cv.bn_b = pf(base + 2); cv.bn_g = pf(base + 3);
    for (int h = 0; h < 4; ++h) {
      int hb = base + 4 + h * 12;
      cv.h[h] = { pf(hb+0), pf(hb+1), pf(hb+2),  pf(hb+3),
                  pf(hb+4), pf(hb+5), pf(hb+6),  pf(hb+7),
                  pf(hb+8), pf(hb+9), pf(hb+10), pf(hb+11) };
    }
    return cv;
  };
  Conv l1[2] = { mk_conv(8),   mk_conv(8 + 52)   };
  Conv l2[2] = { mk_conv(112), mk_conv(112 + 52) };

  // ---------------- workspace carve-up (f32 elements) -----------------------
  float* w = (float*)d_ws;
  float* XV   = w;                    w += (long long)NN * 128;   // node feats
  float* EF   = w;                    w += (long long)EE * 128;   // edge feats
  float* TF   = w;                    w += (long long)TT * 128;   // tri feats
  float* KV   = w;                    w += (long long)EE * 256;   // [Kv|Vv]
  float* KE   = w;                    w += (long long)TT * 256;   // [Ke|Ve] / h
  float* GATE = w;                    w += (long long)TT * 256;
  float* MB   = w;                    w += (long long)TT * 256;   // m / t
  float* HV   = w;                    w += (long long)EE * 512;   // head concat
  float* OUTB = w;                    w += (long long)EE * 128;   // conv out
  float* S    = w;                    w += 128;
  float* S2   = w;                    w += 128;
  float* P1   = w;                    w += 64 * 128;
  float* C1   = w;                    w += 64;
  float* P2   = w;                    w += 64 * 128;
  float* C2   = w;                    w += 64;
  float* F    = w;                    w += 64 * 256;
  float* G1   = w;                    w += 64 * 128;
  float* G2   = w;                    w += 64 * 128;
  // total ~1.14 GB

  // ---------------- input projections --------------------------------------
  gemm(stream, x,    NN, 92,  92,  atom_W, 128, atom_b, XV, 128, 128);
  gemm(stream, edis, EE, 376, 376, edge_W, 128, edge_b, EF, 128, 128);
  gemm(stream, tdis, TT, 516, 516, tri_W,  128, tri_b,  TF, 128, 128);

  // ---------------- layer2: convs on edge features (graph = triangles) -----
  const int* tsrc = tidx;           const int* tdst = tidx + TT;
  for (int l = 0; l < 2; ++l)
    run_conv(stream, EF, EE, tsrc, tdst, TF, TT, l2[l],
             KV, KE, GATE, MB, HV, OUTB, S, S2);

  // ---------------- layer1: convs on node features (graph = edges) ---------
  const int* esrc = eidx;           const int* edst = eidx + EE;
  for (int l = 0; l < 2; ++l)
    run_conv(stream, XV, NN, esrc, edst, EF, EE, l1[l],
             KV, KE, GATE, MB, HV, OUTB, S, S2);

  // ---------------- mean pooling + head MLP --------------------------------
  zero(stream, P1, 64 * 128); zero(stream, C1, 64);
  zero(stream, P2, 64 * 128); zero(stream, C2, 64);
  pool_scatter_kernel<<<dim3(NN), dim3(128), 0, stream>>>(XV, xb,   P1, C1);
  pool_scatter_kernel<<<dim3(EE), dim3(128), 0, stream>>>(EF, ebat, P2, C2);
  pool_final_kernel<<<dim3(64), dim3(256), 0, stream>>>(P1, C1, P2, C2, F);

  gemm(stream, F,  64, 256, 256, fc_W,  128, fc_b,  G1, 128, 128,
       nullptr, 0, 2 /*lrelu*/);
  gemm(stream, G1, 64, 128, 128, fc2_W, 128, fc2_b, G2, 128, 128,
       nullptr, 0, 2 /*lrelu*/);
  out_dot_kernel<<<dim3(64), dim3(128), 0, stream>>>(G2, out_W, out_b,
                                                     (float*)d_out);
}
